// CausalSelfAttention_11072425689962
// MI455X (gfx1250) — matrix operand order repaired and checked by reading, not compiled, not run on hardware
//
#include <hip/hip_runtime.h>

// ---------------------------------------------------------------------------
// CDNA5 (gfx1250, wave32) causal self-attention block:
//   f32->bf16 pre-convert -> qkv GEMM (+rmsnorm+rope+gain fused)
//   -> flash attention -> out proj
// All matmuls via v_wmma_f32_16x16x32_bf16 (f32 accumulate).
// Fragments are assembled with zero VALU: contiguous bf16 runs bit-cast
// straight from b128 loads into the WMMA operand vectors.
// ---------------------------------------------------------------------------

typedef __attribute__((ext_vector_type(16))) __bf16 v16bf;
typedef __attribute__((ext_vector_type(8)))  float  v8f;

#define S_LEN   4096
#define DIM     1024
#define NH      16
#define NKV     4
#define HD      64

__device__ __forceinline__ unsigned short f2bfbits(float f) {
    unsigned u = __builtin_bit_cast(unsigned, f);
    u += 0x7FFFu + ((u >> 16) & 1u);          // round-to-nearest-even
    return (unsigned short)(u >> 16);
}
__device__ __forceinline__ unsigned pack2bf(float lo, float hi) {
    return (unsigned)f2bfbits(lo) | ((unsigned)f2bfbits(hi) << 16);
}

__device__ __forceinline__ v8f wmma_bf16(v16bf a, v16bf b, v8f c) {
    return __builtin_amdgcn_wmma_f32_16x16x32_bf16(
        false, a, false, b, (short)0, c, false, false);
}

// Fragment = two contiguous 8x bf16 (16B) runs, reinterpreted in place.
// v16bf element i lives at bytes [2i,2i+1] => memory order == lane order.
struct frag_bits { uint4 lo, hi; };
__device__ __forceinline__ v16bf frag_from2(const unsigned short* p0,
                                            const unsigned short* p1) {
    frag_bits r;
    r.lo = *(const uint4*)p0;
    r.hi = *(const uint4*)p1;
    return __builtin_bit_cast(v16bf, r);
}
__device__ __forceinline__ v16bf frag_from32(const unsigned short* p) {
    return frag_from2(p, p + 8);
}

// ---------------------------------------------------------------------------
// Kernel 0: fp32 -> bf16 conversion, 8 elements per thread.
// ---------------------------------------------------------------------------
__global__ __launch_bounds__(256) void cvt_kernel(
    const float* __restrict__ in, unsigned short* __restrict__ out, int n8)
{
    const int i = blockIdx.x * 256 + threadIdx.x;
    if (i >= n8) return;
    const float4* p = (const float4*)in + 2 * (size_t)i;
    const float4 f0 = p[0], f1 = p[1];
    uint4 o;
    o.x = pack2bf(f0.x, f0.y);
    o.y = pack2bf(f0.z, f0.w);
    o.z = pack2bf(f1.x, f1.y);
    o.w = pack2bf(f1.z, f1.w);
    *((uint4*)out + i) = o;
}

// ---------------------------------------------------------------------------
// Kernel 1: fused QKV projection + per-head RMSNorm + RoPE + q_gain.
// grid = (S/64, NH + 2*NKV), block = 128 (4 waves, one 16x64 tile each).
// V is stored TRANSPOSED (Vt[kvh][d][s]) so attention's PV B-frags are
// contiguous global loads.
// ---------------------------------------------------------------------------
__global__ __launch_bounds__(128) void qkv_kernel(
    const unsigned short* __restrict__ xb,
    const unsigned short* __restrict__ Wqb,
    const unsigned short* __restrict__ Wkb,
    const unsigned short* __restrict__ Wvb,
    const float* __restrict__ qgain,
    unsigned short* __restrict__ Qb,   // [S][NH*HD]
    unsigned short* __restrict__ Kb,   // [NKV][S][HD]
    unsigned short* __restrict__ Vt)   // [NKV][HD][S]  (transposed)
{
    const int lane = threadIdx.x & 31;
    const int wv   = threadIdx.x >> 5;
    const int s0   = (blockIdx.x * 4 + wv) * 16;
    const int hh   = blockIdx.y;
    const int hi   = lane >> 4;
    const int ln   = lane & 15;

    const unsigned short* W; int wrow0, kind, hd;
    if (hh < NH)            { W = Wqb; wrow0 = hh * HD;        kind = 0; hd = hh; }
    else if (hh < NH + NKV) { W = Wkb; wrow0 = (hh - NH) * HD; kind = 1; hd = hh - NH; }
    else                    { W = Wvb; wrow0 = (hh - NH - NKV) * HD; kind = 2; hd = hh - NH - NKV; }

    v8f acc[4] = {};                       // 16 rows x 64 cols (4 n-tiles)

    const unsigned short* pa_base = xb + (size_t)(s0 + ln) * DIM;
    for (int kk = 0; kk < DIM; kk += 32) {
        // A (m = ln): elems 0-15 <- k = hi*16+i
        const v16bf a = frag_from32(pa_base + kk + hi * 16);
#pragma unroll
        for (int t = 0; t < 4; ++t) {      // B[k][n] = W[wrow0+16t+n][kk+k]
            const v16bf b = frag_from32(
                W + (size_t)(wrow0 + 16 * t + ln) * DIM + kk + hi * 16);
            acc[t] = wmma_bf16(a, b, acc[t]);
        }
    }

    if (kind != 2) {
        // RMS-norm over the 64 values of each row (16 lanes x 4 tiles)
        float r[8];
#pragma unroll
        for (int j = 0; j < 8; ++j) {
            float ss = acc[0][j]*acc[0][j] + acc[1][j]*acc[1][j]
                     + acc[2][j]*acc[2][j] + acc[3][j]*acc[3][j];
            ss += __shfl_xor(ss, 1, 32);
            ss += __shfl_xor(ss, 2, 32);
            ss += __shfl_xor(ss, 4, 32);
            ss += __shfl_xor(ss, 8, 32);
            r[j] = rsqrtf(ss * (1.0f / 64.0f) + 1.1920929e-07f);
        }
        const float g = (kind == 0) ? qgain[hd] : 1.0f;
        // RoPE: col d pairs with d+32 (tile t with tile t+2)
#pragma unroll
        for (int t = 0; t < 2; ++t) {
            const int d = ln + 16 * t;                       // 0..31
            const float invf = __expf(-(float)d * 0.28782313662425576f); // 10000^(-d/32)
#pragma unroll
            for (int j = 0; j < 8; ++j) {
                const float ang = (float)(s0 + j + 8 * hi) * invf;
                const float c = __cosf(ang), sn = __sinf(ang);
                const float a1 = acc[t][j]     * r[j];
                const float a2 = acc[t + 2][j] * r[j];
                acc[t][j]     = ( a1 * c + a2 * sn) * g;
                acc[t + 2][j] = (-a1 * sn + a2 * c) * g;
            }
        }
    }

#pragma unroll
    for (int t = 0; t < 4; ++t) {
#pragma unroll
        for (int j = 0; j < 8; ++j) {
            const int m = j + 8 * hi;
            const int d = ln + 16 * t;
            const unsigned short h = f2bfbits(acc[t][j]);
            if (kind == 0)
                Qb[(size_t)(s0 + m) * DIM + hd * HD + d] = h;
            else if (kind == 1)
                Kb[(size_t)hd * S_LEN * HD + (size_t)(s0 + m) * HD + d] = h;
            else  // V transposed: Vt[hd][d][s]
                Vt[((size_t)hd * HD + d) * S_LEN + (s0 + m)] = h;
        }
    }
}

// ---------------------------------------------------------------------------
// Kernel 2: flash attention (causal, GQA). grid = (S/16, NH), block = 32.
// K and V^T fragments load straight from global (contiguous per lane).
// Only P (16x32) bounces through LDS for its cross-lane transpose.
// ---------------------------------------------------------------------------
__global__ __launch_bounds__(32) void attn_kernel(
    const unsigned short* __restrict__ Qb,
    const unsigned short* __restrict__ Kb,
    const unsigned short* __restrict__ Vt,
    unsigned short* __restrict__ Ob)    // [S][NH*HD]
{
    __shared__ __align__(16) unsigned short ldsP[16 * 32];

    const int lane = threadIdx.x;
    const int hi = lane >> 4, ln = lane & 15;
    const int q0  = blockIdx.x * 16;
    const int h   = blockIdx.y;
    const int kvh = h >> 2;
    const unsigned short* K = Kb + (size_t)kvh * S_LEN * HD;
    const unsigned short* V = Vt + (size_t)kvh * HD * S_LEN;

    // Q A-frags for d in [0,32) and [32,64)
    v16bf aq[2];
    {
        const unsigned short* pq = Qb + (size_t)(q0 + ln) * DIM + h * HD;
#pragma unroll
        for (int s2 = 0; s2 < 2; ++s2)
            aq[s2] = frag_from32(pq + s2 * 32 + hi * 16);
    }

    float mr[8], lr[8];
#pragma unroll
    for (int j = 0; j < 8; ++j) { mr[j] = -3.0e38f; lr[j] = 0.0f; }
    v8f o[4] = {};
    const float scale = 0.125f;            // 1/sqrt(64)

    for (int kv0 = 0; kv0 <= q0 + 15; kv0 += 32) {
        // S = Q K^T   (two 16x16 n-tiles, K-depth 64 = 2 wmma each)
        v8f sacc[2] = {};
#pragma unroll
        for (int t = 0; t < 2; ++t) {
#pragma unroll
            for (int s2 = 0; s2 < 2; ++s2) {
                // B[k=d][n] = K[kv0+16t+n][s2*32 + k]
                const v16bf b = frag_from32(
                    K + (size_t)(kv0 + 16 * t + ln) * HD + s2 * 32 + hi * 16);
                sacc[t] = wmma_bf16(aq[s2], b, sacc[t]);
            }
        }

        // online softmax update (row stats across 16-lane halves)
        float p0[8], p1[8];
#pragma unroll
        for (int j = 0; j < 8; ++j) {
            const int qg = q0 + j + 8 * hi;
            float v0 = sacc[0][j] * scale; if (kv0 + ln      > qg) v0 = -3.0e38f;
            float v1 = sacc[1][j] * scale; if (kv0 + 16 + ln > qg) v1 = -3.0e38f;
            float mx = fmaxf(v0, v1);
            mx = fmaxf(mx, __shfl_xor(mx, 1, 32));
            mx = fmaxf(mx, __shfl_xor(mx, 2, 32));
            mx = fmaxf(mx, __shfl_xor(mx, 4, 32));
            mx = fmaxf(mx, __shfl_xor(mx, 8, 32));
            const float mn    = fmaxf(mr[j], mx);
            const float alpha = __expf(mr[j] - mn);
            const float e0 = __expf(v0 - mn);
            const float e1 = __expf(v1 - mn);
            float rs = e0 + e1;
            rs += __shfl_xor(rs, 1, 32);
            rs += __shfl_xor(rs, 2, 32);
            rs += __shfl_xor(rs, 4, 32);
            rs += __shfl_xor(rs, 8, 32);
            lr[j] = lr[j] * alpha + rs;
            mr[j] = mn;
            p0[j] = e0; p1[j] = e1;
#pragma unroll
            for (int dt = 0; dt < 4; ++dt) o[dt][j] *= alpha;
        }

        // P (C-layout) -> LDS row-major [16][32] bf16 (packed 32b stores
        // would straddle lanes; keep 16b stores, it's only 16/lane)
#pragma unroll
        for (int j = 0; j < 8; ++j) {
            ldsP[(j + 8 * hi) * 32 +      ln] = f2bfbits(p0[j]);
            ldsP[(j + 8 * hi) * 32 + 16 + ln] = f2bfbits(p1[j]);
        }
        __syncthreads();

        // P as A-frag (two contiguous ds_load_b128)
        const v16bf ap = frag_from32(&ldsP[ln * 32 + hi * 16]);

        // O += P * V : B[k][n] = V^T[dt*16+n][kv0+k], contiguous per lane
#pragma unroll
        for (int dt = 0; dt < 4; ++dt) {
            const v16bf bv = frag_from32(
                V + (size_t)(dt * 16 + ln) * S_LEN + kv0 + hi * 16);
            o[dt] = wmma_bf16(ap, bv, o[dt]);
        }
        __syncthreads();
    }

#pragma unroll
    for (int dt = 0; dt < 4; ++dt) {
#pragma unroll
        for (int j = 0; j < 8; ++j) {
            const int m = j + 8 * hi;
            Ob[(size_t)(q0 + m) * DIM + h * HD + dt * 16 + ln] =
                f2bfbits(o[dt][j] / lr[j]);
        }
    }
}

// ---------------------------------------------------------------------------
// Kernel 3: output projection y = Ob @ Wproj^T (fp32 out).
// grid = (S/64, DIM/64), block = 128 (4 waves, 16x64 tile each).
// ---------------------------------------------------------------------------
__global__ __launch_bounds__(128) void proj_kernel(
    const unsigned short* __restrict__ Ob,
    const unsigned short* __restrict__ Wpb,
    float* __restrict__ y)
{
    const int lane = threadIdx.x & 31;
    const int wv   = threadIdx.x >> 5;
    const int s0   = (blockIdx.x * 4 + wv) * 16;
    const int n0   = blockIdx.y * 64;
    const int hi   = lane >> 4, ln = lane & 15;

    v8f acc[4] = {};
    const unsigned short* pa_base = Ob + (size_t)(s0 + ln) * DIM;
    for (int kk = 0; kk < DIM; kk += 32) {
        const v16bf a = frag_from32(pa_base + kk + hi * 16);
#pragma unroll
        for (int t = 0; t < 4; ++t) {
            const v16bf b = frag_from32(
                Wpb + (size_t)(n0 + 16 * t + ln) * DIM + kk + hi * 16);
            acc[t] = wmma_bf16(a, b, acc[t]);
        }
    }
#pragma unroll
    for (int t = 0; t < 4; ++t)
#pragma unroll
        for (int j = 0; j < 8; ++j)
            y[(size_t)(s0 + j + 8 * hi) * DIM + n0 + 16 * t + ln] = acc[t][j];
}

// ---------------------------------------------------------------------------
extern "C" void kernel_launch(void* const* d_in, const int* in_sizes, int n_in,
                              void* d_out, int out_size, void* d_ws, size_t ws_size,
                              hipStream_t stream)
{
    const float* x  = (const float*)d_in[0];
    const float* Wq = (const float*)d_in[1];
    const float* Wk = (const float*)d_in[2];
    const float* Wv = (const float*)d_in[3];
    const float* Wp = (const float*)d_in[4];
    const float* qg = (const float*)d_in[5];

    char* ws = (char*)d_ws;
    size_t off = 0;
    auto carve = [&](size_t elems) {
        unsigned short* p = (unsigned short*)(ws + off);
        off += elems * 2;
        return p;
    };
    unsigned short* xb  = carve((size_t)S_LEN * DIM);          // 8 MiB
    unsigned short* Wqb = carve((size_t)DIM * DIM);            // 2 MiB
    unsigned short* Wkb = carve((size_t)NKV * HD * DIM);       // 0.5 MiB
    unsigned short* Wvb = carve((size_t)NKV * HD * DIM);       // 0.5 MiB
    unsigned short* Wpb = carve((size_t)DIM * DIM);            // 2 MiB
    unsigned short* Qb  = carve((size_t)S_LEN * DIM);          // 8 MiB
    unsigned short* Kb  = carve((size_t)NKV * S_LEN * HD);     // 2 MiB
    unsigned short* Vt  = carve((size_t)NKV * HD * S_LEN);     // 2 MiB
    unsigned short* Ob  = carve((size_t)S_LEN * DIM);          // 8 MiB

    // fp32 -> bf16 pre-conversion (amortized once, not per GEMM tile)
    auto cvt = [&](const float* src, unsigned short* dst, size_t n) {
        const int n8 = (int)(n / 8);
        cvt_kernel<<<(n8 + 255) / 256, 256, 0, stream>>>(src, dst, n8);
    };
    cvt(x,  xb,  (size_t)S_LEN * DIM);
    cvt(Wq, Wqb, (size_t)DIM * DIM);
    cvt(Wk, Wkb, (size_t)NKV * HD * DIM);
    cvt(Wv, Wvb, (size_t)NKV * HD * DIM);
    cvt(Wp, Wpb, (size_t)DIM * DIM);

    qkv_kernel <<<dim3(S_LEN / 64, NH + 2 * NKV), 128, 0, stream>>>(
        xb, Wqb, Wkb, Wvb, qg, Qb, Kb, Vt);
    attn_kernel<<<dim3(S_LEN / 16, NH),            32, 0, stream>>>(
        Qb, Kb, Vt, Ob);
    proj_kernel<<<dim3(S_LEN / 64, DIM / 64),     128, 0, stream>>>(
        Ob, Wpb, (float*)d_out);
}
